// OptFC1EarlyTriton_48679159333253
// MI455X (gfx1250) — matrix operand order repaired and checked by reading, not compiled, not run on hardware
//
#include <hip/hip_runtime.h>

typedef __attribute__((ext_vector_type(16))) __bf16 v16bf;
typedef __attribute__((ext_vector_type(8)))  float  v8f;

#define TILE_M   128
#define TILE_N   128
#define TILE_K   32
#define LDS_KP   40   // padded bf16 row stride (elements) -> 80B, conflict-spread
#define LDS_KP0  33   // padded f32 stats row stride (elements)
#define EARLY_IT 32
#define TILE_HALFS (TILE_M * LDS_KP)          // halfs per LDS matrix tile
#define STAGE_HALFS (4 * TILE_HALFS)          // Ah,Al,Bh,Bl per stage

#if __has_builtin(__builtin_amdgcn_global_load_async_to_lds_b128) && \
    __has_builtin(__builtin_amdgcn_s_wait_asynccnt)
#define USE_ASYNC_LDS 1
typedef int vi4 __attribute__((vector_size(16)));          // native clang vector
typedef __attribute__((address_space(1))) vi4 vi4_g;       // global (AS1)
typedef __attribute__((address_space(3))) vi4 vi4_l;       // LDS (AS3)
#define GAS(p) ((vi4_g*)(void*)(p))
#define LAS(p) ((vi4_l*)(void*)(p))
#else
#define USE_ASYNC_LDS 0
#endif

__device__ __forceinline__ unsigned short f2bf(float f) {
    unsigned u = __float_as_uint(f);
    u += 0x7FFFu + ((u >> 16) & 1u);           // round-to-nearest-even
    return (unsigned short)(u >> 16);
}
__device__ __forceinline__ float bf2f(unsigned short s) {
    return __uint_as_float(((unsigned)s) << 16);
}

union FragU { v16bf v; int4 q[2]; };

// ---- one-time fp32 -> (bf16 hi, bf16 lo) split, bandwidth-bound ------------
__global__ __launch_bounds__(256)
void convert_split(const float* __restrict__ src, unsigned short* __restrict__ hi,
                   unsigned short* __restrict__ lo, size_t n4) {
    size_t stride = (size_t)gridDim.x * 256;
    for (size_t i = (size_t)blockIdx.x * 256 + threadIdx.x; i < n4; i += stride) {
        float4 v = ((const float4*)src)[i];
        ushort4 h, l;
        h.x = f2bf(v.x); l.x = f2bf(v.x - bf2f(h.x));
        h.y = f2bf(v.y); l.y = f2bf(v.y - bf2f(h.y));
        h.z = f2bf(v.z); l.z = f2bf(v.z - bf2f(h.z));
        h.w = f2bf(v.w); l.w = f2bf(v.w - bf2f(h.w));
        ((ushort4*)hi)[i] = h;
        ((ushort4*)lo)[i] = l;
    }
}

template<bool PRE>
__global__ __launch_bounds__(256)
void fc1_early_wmma(const float* __restrict__ x, const float* __restrict__ wgt,
                    const float* __restrict__ bias,
                    const unsigned short* __restrict__ xh, const unsigned short* __restrict__ xl,
                    const unsigned short* __restrict__ wh, const unsigned short* __restrict__ wl,
                    float* __restrict__ out, int M, int N, int K) {
    // two LDS stages for double buffering (PRE path); fallback uses stage 0 only
    __shared__ __align__(16) unsigned short smem[2 * STAGE_HALFS];   // 80 KB
    float* A0 = (float*)smem;                  // stats phase reuses stage 0
    float* B0 = A0 + TILE_M * LDS_KP0;         // 33 KB < 40 KB

    const int tid   = threadIdx.x;
    const int lane  = tid & 31;
    const int half  = lane >> 4;
    const int lm    = lane & 15;
    const int wave  = tid >> 5;
    const int wm    = wave & 3;    // 4 wave rows  -> 32 M each
    const int wn    = wave >> 2;   // 2 wave cols  -> 64 N each
    const int blockN = blockIdx.x * TILE_N;
    const int blockM = blockIdx.y * TILE_M;

    // ---------------- Phase 1: fp32 stats over first 32 K-columns ----------
    #pragma unroll
    for (int i = 0; i < 4; ++i) {
        int idx = tid + i * 256;               // 0..1023
        int row = idx >> 3, c4 = (idx & 7) * 4;
        float4 va = *(const float4*)(x   + (size_t)(blockM + row) * K + c4);
        float4 vb = *(const float4*)(wgt + (size_t)(blockN + row) * K + c4);
        float* ap = A0 + row * LDS_KP0 + c4;
        ap[0] = va.x; ap[1] = va.y; ap[2] = va.z; ap[3] = va.w;
        float* bp = B0 + row * LDS_KP0 + c4;
        bp[0] = vb.x; bp[1] = vb.y; bp[2] = vb.z; bp[3] = vb.w;
    }
    __syncthreads();

    float y1a[2][8][4], qa[2][8][4];
    #pragma unroll
    for (int tm = 0; tm < 2; ++tm)
        #pragma unroll
        for (int r = 0; r < 8; ++r)
            #pragma unroll
            for (int tn = 0; tn < 4; ++tn) { y1a[tm][r][tn] = 0.f; qa[tm][r][tn] = 0.f; }

    for (int k = 0; k < EARLY_IT; ++k) {
        float bv[4], b2[4];
        #pragma unroll
        for (int tn = 0; tn < 4; ++tn) {
            float b = B0[(wn * 64 + tn * 16 + lm) * LDS_KP0 + k];
            bv[tn] = b; b2[tn] = b * b;
        }
        #pragma unroll
        for (int tm = 0; tm < 2; ++tm)
            #pragma unroll
            for (int r = 0; r < 8; ++r) {
                float a  = A0[(wm * 32 + tm * 16 + half * 8 + r) * LDS_KP0 + k];
                float a2 = a * a;
                #pragma unroll
                for (int tn = 0; tn < 4; ++tn) {
                    y1a[tm][r][tn] = fmaf(a,  bv[tn], y1a[tm][r][tn]);
                    qa [tm][r][tn] = fmaf(a2, b2[tn], qa [tm][r][tn]);
                }
            }
    }

    unsigned long long pm = 0ull;              // per-lane pass mask, 64 outputs
    {
        const float inv = 1.0f / (float)EARLY_IT;
        const float bound = -2.053748910631823f;
        #pragma unroll
        for (int tm = 0; tm < 2; ++tm)
            #pragma unroll
            for (int tn = 0; tn < 4; ++tn)
                #pragma unroll
                for (int r = 0; r < 8; ++r) {
                    float mean  = y1a[tm][r][tn] * inv;
                    float var   = qa[tm][r][tn] * inv - mean * mean;
                    float denom = sqrtf(fmaxf(var * inv, 1e-12f));
                    float stat  = mean / denom;
                    if (stat < bound)
                        pm |= 1ull << ((tm * 4 + tn) * 8 + r);
                }
    }
    __syncthreads();   // stats LDS dead; reuse for bf16 tiles

    // ---------------- Phase 2: full-K split-bf16 WMMA GEMM -----------------
    v8f acc[2][4];
    #pragma unroll
    for (int tm = 0; tm < 2; ++tm)
        #pragma unroll
        for (int tn = 0; tn < 4; ++tn) {
            v8f z = {0.f, 0.f, 0.f, 0.f, 0.f, 0.f, 0.f, 0.f};
            acc[tm][tn] = z;
        }

    // fragment load + 24 WMMAs for the tile staged in LDS stage s
    auto compute = [&](int s) {
        const unsigned short* Ah = smem + s * STAGE_HALFS;
        const unsigned short* Al = Ah + TILE_HALFS;
        const unsigned short* Bh = Al + TILE_HALFS;
        const unsigned short* Bl = Bh + TILE_HALFS;
        FragU bh[4], bl[4];
        #pragma unroll
        for (int t = 0; t < 4; ++t) {          // B frags: two 16B K-chunks per lane
            int n = wn * 64 + t * 16 + lm;
            bh[t].q[0] = *(const int4*)(Bh + n * LDS_KP + 8 * half);
            bh[t].q[1] = *(const int4*)(Bh + n * LDS_KP + 16 + 8 * half);
            bl[t].q[0] = *(const int4*)(Bl + n * LDS_KP + 8 * half);
            bl[t].q[1] = *(const int4*)(Bl + n * LDS_KP + 16 + 8 * half);
        }
        #pragma unroll
        for (int tm = 0; tm < 2; ++tm) {
            int m = wm * 32 + tm * 16 + lm;
            FragU ah, al;                      // A frags: two 16B K-chunks per lane
            ah.q[0] = *(const int4*)(Ah + m * LDS_KP + 8 * half);
            ah.q[1] = *(const int4*)(Ah + m * LDS_KP + 16 + 8 * half);
            al.q[0] = *(const int4*)(Al + m * LDS_KP + 8 * half);
            al.q[1] = *(const int4*)(Al + m * LDS_KP + 16 + 8 * half);
            #pragma unroll
            for (int tn = 0; tn < 4; ++tn) {
                acc[tm][tn] = __builtin_amdgcn_wmma_f32_16x16x32_bf16(
                    false, ah.v, false, bh[tn].v, (short)0, acc[tm][tn], false, false);
                acc[tm][tn] = __builtin_amdgcn_wmma_f32_16x16x32_bf16(
                    false, ah.v, false, bl[tn].v, (short)0, acc[tm][tn], false, false);
                acc[tm][tn] = __builtin_amdgcn_wmma_f32_16x16x32_bf16(
                    false, al.v, false, bh[tn].v, (short)0, acc[tm][tn], false, false);
            }
        }
    };

    if (PRE) {
#if USE_ASYNC_LDS
        // CDNA5 async DMA: global -> LDS with no VGPR staging (ASYNCcnt)
        auto stage_copy = [&](int kt, int s) {
            unsigned short* D = smem + s * STAGE_HALFS;
            #pragma unroll
            for (int j = 0; j < 2; ++j) {
                int c = tid + j * 256;         // 512 16B-chunks per 128x32 tile
                int row = c >> 2, off = (c & 3) * 8;
                size_t ax = (size_t)(blockM + row) * K + kt + off;
                size_t aw = (size_t)(blockN + row) * K + kt + off;
                int dst = row * LDS_KP + off;
                __builtin_amdgcn_global_load_async_to_lds_b128(
                    GAS(xh + ax), LAS(D + 0 * TILE_HALFS + dst), 0, 0);
                __builtin_amdgcn_global_load_async_to_lds_b128(
                    GAS(xl + ax), LAS(D + 1 * TILE_HALFS + dst), 0, 0);
                __builtin_amdgcn_global_load_async_to_lds_b128(
                    GAS(wh + aw), LAS(D + 2 * TILE_HALFS + dst), 0, 0);
                __builtin_amdgcn_global_load_async_to_lds_b128(
                    GAS(wl + aw), LAS(D + 3 * TILE_HALFS + dst), 0, 0);
            }
        };
        stage_copy(0, 0);
        __builtin_amdgcn_s_wait_asynccnt(0);
        __syncthreads();
        for (int kt = 0; kt < K; kt += 2 * TILE_K) {   // K % 64 == 0
            if (kt + TILE_K < K) stage_copy(kt + TILE_K, 1);   // DMA overlaps WMMAs
            compute(0);
            __builtin_amdgcn_s_wait_asynccnt(0);
            __syncthreads();
            if (kt + 2 * TILE_K < K) stage_copy(kt + 2 * TILE_K, 0);
            compute(1);
            __builtin_amdgcn_s_wait_asynccnt(0);
            __syncthreads();
        }
#else
        // fallback: fused load->store, short register liveness (no spills)
        auto stage_copy = [&](int kt, int s) {
            unsigned short* D = smem + s * STAGE_HALFS;
            #pragma unroll
            for (int j = 0; j < 2; ++j) {
                int c = tid + j * 256;
                int row = c >> 2, off = (c & 3) * 8;
                size_t ax = (size_t)(blockM + row) * K + kt + off;
                size_t aw = (size_t)(blockN + row) * K + kt + off;
                int dst = row * LDS_KP + off;
                int4 a0 = *(const int4*)(xh + ax);
                int4 a1 = *(const int4*)(xl + ax);
                int4 b0 = *(const int4*)(wh + aw);
                int4 b1 = *(const int4*)(wl + aw);
                *(int4*)(D + 0 * TILE_HALFS + dst) = a0;
                *(int4*)(D + 1 * TILE_HALFS + dst) = a1;
                *(int4*)(D + 2 * TILE_HALFS + dst) = b0;
                *(int4*)(D + 3 * TILE_HALFS + dst) = b1;
            }
        };
        stage_copy(0, 0);
        __syncthreads();
        for (int kt = 0; kt < K; kt += 2 * TILE_K) {   // K % 64 == 0
            if (kt + TILE_K < K) stage_copy(kt + TILE_K, 1);
            compute(0);
            __syncthreads();
            if (kt + 2 * TILE_K < K) stage_copy(kt + 2 * TILE_K, 0);
            compute(1);
            __syncthreads();
        }
#endif
    } else {
        // fallback: fused fp32 load + split conversion each iteration (stage 0)
        unsigned short* Ah = smem;
        unsigned short* Al = Ah + TILE_HALFS;
        unsigned short* Bh = Al + TILE_HALFS;
        unsigned short* Bl = Bh + TILE_HALFS;
        for (int kt = 0; kt < K; kt += TILE_K) {
            #pragma unroll
            for (int i = 0; i < 4; ++i) {
                int idx = tid + i * 256;
                int row = idx >> 3, c4 = (idx & 7) * 4;
                float4 va = *(const float4*)(x + (size_t)(blockM + row) * K + kt + c4);
                ushort4 hi, lo;
                hi.x = f2bf(va.x); lo.x = f2bf(va.x - bf2f(hi.x));
                hi.y = f2bf(va.y); lo.y = f2bf(va.y - bf2f(hi.y));
                hi.z = f2bf(va.z); lo.z = f2bf(va.z - bf2f(hi.z));
                hi.w = f2bf(va.w); lo.w = f2bf(va.w - bf2f(hi.w));
                *(ushort4*)(Ah + row * LDS_KP + c4) = hi;
                *(ushort4*)(Al + row * LDS_KP + c4) = lo;
                float4 vb = *(const float4*)(wgt + (size_t)(blockN + row) * K + kt + c4);
                hi.x = f2bf(vb.x); lo.x = f2bf(vb.x - bf2f(hi.x));
                hi.y = f2bf(vb.y); lo.y = f2bf(vb.y - bf2f(hi.y));
                hi.z = f2bf(vb.z); lo.z = f2bf(vb.z - bf2f(hi.z));
                hi.w = f2bf(vb.w); lo.w = f2bf(vb.w - bf2f(hi.w));
                *(ushort4*)(Bh + row * LDS_KP + c4) = hi;
                *(ushort4*)(Bl + row * LDS_KP + c4) = lo;
            }
            __syncthreads();
            compute(0);
            __syncthreads();
        }
    }

    // ---------------- Epilogue: bias, mask, store ---------------------------
    #pragma unroll
    for (int tm = 0; tm < 2; ++tm)
        #pragma unroll
        for (int tn = 0; tn < 4; ++tn) {
            int ng = blockN + wn * 64 + tn * 16 + lm;
            float bsv = bias[ng];
            int tidx = tm * 4 + tn;
            #pragma unroll
            for (int r = 0; r < 8; ++r) {
                int mg = blockM + wm * 32 + tm * 16 + half * 8 + r;
                float v = acc[tm][tn][r] + bsv;
                if ((pm >> (tidx * 8 + r)) & 1ull) v = 0.0f;
                out[(size_t)mg * N + ng] = v;
            }
        }
}

extern "C" void kernel_launch(void* const* d_in, const int* in_sizes, int n_in,
                              void* d_out, int out_size, void* d_ws, size_t ws_size,
                              hipStream_t stream) {
    const float* x    = (const float*)d_in[0];
    const float* wgt  = (const float*)d_in[1];
    const float* bias = (const float*)d_in[2];
    float* out = (float*)d_out;
    int N = in_sizes[2];
    int K = in_sizes[1] / N;
    int M = in_sizes[0] / K;
    size_t elemsX = (size_t)M * K, elemsW = (size_t)N * K;
    size_t need = (elemsX + elemsW) * 2 * sizeof(unsigned short); // hi+lo
    dim3 grid(N / TILE_N, M / TILE_M), blk(256);

    if (ws_size >= need) {
        unsigned short* xh = (unsigned short*)d_ws;
        unsigned short* xl = xh + elemsX;
        unsigned short* wh = xl + elemsX;
        unsigned short* wl = wh + elemsW;
        convert_split<<<dim3(2048), blk, 0, stream>>>(x,   xh, xl, elemsX / 4);
        convert_split<<<dim3(2048), blk, 0, stream>>>(wgt, wh, wl, elemsW / 4);
        fc1_early_wmma<true><<<grid, blk, 0, stream>>>(x, wgt, bias, xh, xl, wh, wl,
                                                       out, M, N, K);
    } else {
        fc1_early_wmma<false><<<grid, blk, 0, stream>>>(x, wgt, bias,
                                                        nullptr, nullptr, nullptr, nullptr,
                                                        out, M, N, K);
    }
}
